// All_Concat_model_new_81243601371615
// MI455X (gfx1250) — matrix operand order repaired and checked by reading, not compiled, not run on hardware
//
#include <hip/hip_runtime.h>
#include <hip/hip_bf16.h>

typedef __attribute__((ext_vector_type(16))) __bf16 v16bf;
typedef __attribute__((ext_vector_type(8)))  float  v8f;
typedef int vint4 __attribute__((vector_size(16)));

#define N_NODES  100000
#define N_EDGES  500000
#define N_GRAPHS 2048
#define DIMF     128
#define OUTD     64
#define FIND     2
#define KDIM     128
#define BN_EPS   1e-5f

#if defined(__has_builtin)
#if __has_builtin(__builtin_amdgcn_global_load_async_to_lds_b128) && \
    __has_builtin(__builtin_amdgcn_s_wait_asynccnt)
#define HAVE_ASYNC_LDS 1
#endif
#endif

__device__ __forceinline__ float4 ld4(const float* p) { return *(const float4*)p; }

// ---------------------------------------------------------------------------
// Pack fp32 weight W[K=128, Nc] into bf16 "B-fragment order":
//   P[((kbi*TILES_N + tn)*32 + lane)*16 + e] = bf16( W[kbi*32 + (lane>>4)*16 + e][tn*16 + (lane&15)] )
// so each lane's 16-element WMMA B fragment is one contiguous 32-byte chunk.
// ---------------------------------------------------------------------------
__global__ void pack_weights_bf16(const float* __restrict__ W, unsigned short* __restrict__ P,
                                  int Nc)
{
    const int idx = blockIdx.x * blockDim.x + threadIdx.x;
    if (idx >= KDIM * Nc) return;
    const int e      = idx & 15;
    const int lane   = (idx >> 4) & 31;
    const int tile   = idx >> 9;               // kbi*tilesN + tn
    const int tilesN = Nc >> 4;
    const int kbi    = tile / tilesN;
    const int tn     = tile - kbi * tilesN;
    const int k      = kbi * 32 + (lane >> 4) * 16 + e;
    const int n      = tn * 16 + (lane & 15);
    const __bf16 b   = (__bf16)W[(size_t)k * Nc + n];
    unsigned short u;
    __builtin_memcpy(&u, &b, 2);
    P[idx] = u;
}

// ---------------------------------------------------------------------------
// Row-strip WMMA GEMM: C[M,Nc] = act(A[M,128] @ W + bias)
//   - one wave per 16-row strip, computing ALL TILES_N column tiles
//     (A fragment reused across TILES_N WMMAs)
//   - packed bf16 weights staged once per block into LDS via async global->LDS
// ACT: 0 = none, 1 = relu.  TILES_N = Nc/16.
// ---------------------------------------------------------------------------
template<int ACT, int TILES_N>
__global__ void gemm_wmma_rowstrip(const float* __restrict__ A,
                                   const unsigned short* __restrict__ Wp,
                                   const float* __restrict__ bias,
                                   float* __restrict__ C, int M)
{
    constexpr int Nc    = TILES_N * 16;
    constexpr int KBLKS = KDIM / 32;                       // 4
    __shared__ __align__(32) unsigned short sW[KBLKS * TILES_N * 512];

    // ---- stage packed weights into LDS (all threads participate) ----
    {
        const int nChunks = (KBLKS * TILES_N * 512 * 2) >> 4;   // 16B chunks
#ifdef HAVE_ASYNC_LDS
        for (int c = threadIdx.x; c < nChunks; c += blockDim.x) {
            __builtin_amdgcn_global_load_async_to_lds_b128(
                (__attribute__((address_space(1))) vint4*)((char*)Wp + (size_t)c * 16),
                (__attribute__((address_space(3))) vint4*)((char*)sW + (size_t)c * 16),
                0, 0);
        }
        __builtin_amdgcn_s_wait_asynccnt(0);
#else
        for (int c = threadIdx.x; c < nChunks; c += blockDim.x)
            ((uint4*)sW)[c] = ((const uint4*)Wp)[c];
#endif
    }
    __syncthreads();

    const int lane  = threadIdx.x & 31;
    const int strip = blockIdx.x * (blockDim.x >> 5) + (threadIdx.x >> 5);
    if (strip * 16 >= M) return;                 // uniform per-wave exit (after barrier)

    const int half = lane >> 4;                  // 0: lanes 0-15, 1: lanes 16-31
    const int l16  = lane & 15;
    const int mrow = strip * 16 + l16;

    v8f acc[TILES_N];
    #pragma unroll
    for (int t = 0; t < TILES_N; ++t) acc[t] = (v8f){0.f,0.f,0.f,0.f,0.f,0.f,0.f,0.f};

    #pragma unroll
    for (int kbi = 0; kbi < KBLKS; ++kbi) {
        // ---- A fragment: 16x32 bf16; lane holds K = kb+half*16+{0..15} ----
        v16bf a;
        {
            float4 a0, a1, a2, a3;
            if (mrow < M) {
                const float* ap = A + (size_t)mrow * KDIM + kbi * 32 + half * 16;
                a0 = ld4(ap);      a1 = ld4(ap + 4);
                a2 = ld4(ap + 8);  a3 = ld4(ap + 12);
            } else {
                a0 = a1 = a2 = a3 = make_float4(0.f, 0.f, 0.f, 0.f);
            }
            a[0]=(__bf16)a0.x;  a[1]=(__bf16)a0.y;  a[2]=(__bf16)a0.z;  a[3]=(__bf16)a0.w;
            a[4]=(__bf16)a1.x;  a[5]=(__bf16)a1.y;  a[6]=(__bf16)a1.z;  a[7]=(__bf16)a1.w;
            a[8]=(__bf16)a2.x;  a[9]=(__bf16)a2.y;  a[10]=(__bf16)a2.z; a[11]=(__bf16)a2.w;
            a[12]=(__bf16)a3.x; a[13]=(__bf16)a3.y; a[14]=(__bf16)a3.z; a[15]=(__bf16)a3.w;
        }
        // ---- sweep all column tiles with this A fragment ----
        const unsigned short* bp = sW + ((size_t)(kbi * TILES_N) * 32 + lane) * 16;
        #pragma unroll
        for (int tn = 0; tn < TILES_N; ++tn) {
            const v16bf b = *(const v16bf*)(bp + (size_t)tn * 512);
            acc[tn] = __builtin_amdgcn_wmma_f32_16x16x32_bf16(false, a, false, b,
                                                              (short)0, acc[tn], false, false);
        }
    }

    // ---- bias + activation + store (C/D layout: VGPR r -> M = r + 8*half) ----
    #pragma unroll
    for (int tn = 0; tn < TILES_N; ++tn) {
        const int ncol = tn * 16 + l16;
        const float bv = bias ? bias[ncol] : 0.f;
        #pragma unroll
        for (int r = 0; r < 8; ++r) {
            const int m = strip * 16 + r + half * 8;
            if (m < M) {
                float v = acc[tn][r] + bv;
                if (ACT == 1) v = fmaxf(v, 0.f);
                C[(size_t)m * Nc + ncol] = v;
            }
        }
    }
}

// ---------------------------------------------------------------------------
// z[dst] += h[src] over all edges; one thread per (edge, float4 chunk)
// ---------------------------------------------------------------------------
__global__ void edge_scatter_add(const int* __restrict__ ei, const float* __restrict__ h,
                                 float* __restrict__ z)
{
    const long long tid = (long long)blockIdx.x * blockDim.x + threadIdx.x;
    if (tid >= (long long)N_EDGES * 32) return;
    const int e  = (int)(tid >> 5);
    const int f4 = (int)(tid & 31) * 4;
    const int s = ei[e];
    const int d = ei[N_EDGES + e];
    const float4 v = ld4(h + (size_t)s * DIMF + f4);
    float* zp = z + (size_t)d * DIMF + f4;
    atomicAdd(zp + 0, v.x);
    atomicAdd(zp + 1, v.y);
    atomicAdd(zp + 2, v.z);
    atomicAdd(zp + 3, v.w);
}

// pooled[batch[n]] += h[n]; one thread per (node, float4 chunk)
__global__ void pool_scatter_add(const int* __restrict__ batch, const float* __restrict__ h,
                                 float* __restrict__ pooled)
{
    const long long tid = (long long)blockIdx.x * blockDim.x + threadIdx.x;
    if (tid >= (long long)N_NODES * 32) return;
    const int n  = (int)(tid >> 5);
    const int f4 = (int)(tid & 31) * 4;
    const int g = batch[n];
    const float4 v = ld4(h + (size_t)n * DIMF + f4);
    float* pp = pooled + (size_t)g * DIMF + f4;
    atomicAdd(pp + 0, v.x);
    atomicAdd(pp + 1, v.y);
    atomicAdd(pp + 2, v.z);
    atomicAdd(pp + 3, v.w);
}

// per-column sum / sum-of-squares over Y[M, DIMF]; blockDim.x == DIMF
__global__ void column_stats(const float* __restrict__ Y, float* __restrict__ csum,
                             float* __restrict__ csq, int M)
{
    const int col = threadIdx.x;
    float s = 0.f, q = 0.f;
    for (int r = blockIdx.x; r < M; r += gridDim.x) {
        const float v = Y[(size_t)r * DIMF + col];
        s += v; q += v * v;
    }
    atomicAdd(&csum[col], s);
    atomicAdd(&csq[col],  q);
}

// Y = relu(gamma * (Y - mu) * rsqrt(var + eps) + beta)
__global__ void bn_relu(float* __restrict__ Y, const float* __restrict__ csum,
                        const float* __restrict__ csq, const float* __restrict__ gamma,
                        const float* __restrict__ beta, int M)
{
    const long long tid = (long long)blockIdx.x * blockDim.x + threadIdx.x;
    if (tid >= (long long)M * DIMF) return;
    const int col = (int)(tid & (DIMF - 1));
    const float invM = 1.0f / (float)M;
    const float mu  = csum[col] * invM;
    const float var = csq[col] * invM - mu * mu;
    const float sc  = gamma[col] * rsqrtf(var + BN_EPS);
    const float v   = (Y[tid] - mu) * sc + beta[col];
    Y[tid] = fmaxf(v, 0.f);
}

__global__ void fill_zero(float* __restrict__ p, long long n)
{
    const long long tid = (long long)blockIdx.x * blockDim.x + threadIdx.x;
    if (tid < n) p[tid] = 0.f;
}

// row-wise log_softmax over 64 columns; one wave per row, 2 elems per lane
__global__ void log_softmax_64(const float* __restrict__ X, float* __restrict__ Y)
{
    const int row  = blockIdx.x * (blockDim.x >> 5) + (threadIdx.x >> 5);
    const int lane = threadIdx.x & 31;
    if (row >= N_GRAPHS) return;
    const float* xp = X + (size_t)row * OUTD;
    const float a = xp[lane], b = xp[lane + 32];
    float mx = fmaxf(a, b);
    for (int o = 16; o > 0; o >>= 1) mx = fmaxf(mx, __shfl_xor(mx, o, 32));
    float se = __expf(a - mx) + __expf(b - mx);
    for (int o = 16; o > 0; o >>= 1) se += __shfl_xor(se, o, 32);
    const float lse = mx + __logf(se);
    float* yp = Y + (size_t)row * OUTD;
    yp[lane]      = a - lse;
    yp[lane + 32] = b - lse;
}

// out[g] = log_softmax( concat(code_emb[g], trans_emb[g]) @ fin_w + fin_b )
__global__ void final_head(const float* __restrict__ code_emb, const float* __restrict__ trans_emb,
                           const float* __restrict__ finw, const float* __restrict__ finb,
                           float* __restrict__ out)
{
    const int g = blockIdx.x * blockDim.x + threadIdx.x;
    if (g >= N_GRAPHS) return;
    float z0 = finb[0], z1 = finb[1];
    for (int k = 0; k < OUTD; ++k) {
        const float c = code_emb[(size_t)g * OUTD + k];
        z0 += c * finw[k * FIND + 0];
        z1 += c * finw[k * FIND + 1];
    }
    for (int k = 0; k < OUTD; ++k) {
        const float t = trans_emb[(size_t)g * OUTD + k];
        z0 += t * finw[(OUTD + k) * FIND + 0];
        z1 += t * finw[(OUTD + k) * FIND + 1];
    }
    const float mx = fmaxf(z0, z1);
    const float lse = mx + __logf(__expf(z0 - mx) + __expf(z1 - mx));
    out[(size_t)g * FIND + 0] = z0 - lse;
    out[(size_t)g * FIND + 1] = z1 - lse;
}

// ---------------------------------------------------------------------------
extern "C" void kernel_launch(void* const* d_in, const int* in_sizes, int n_in,
                              void* d_out, int out_size, void* d_ws, size_t ws_size,
                              hipStream_t stream)
{
    const float* x       = (const float*)d_in[0];
    const int*   ei      = (const int*)  d_in[1];
    const int*   batch   = (const int*)  d_in[2];
    const float* code_x  = (const float*)d_in[3];
    const float* c1_w1   = (const float*)d_in[4];
    const float* c1_b1   = (const float*)d_in[5];
    const float* c1_gam  = (const float*)d_in[6];
    const float* c1_bet  = (const float*)d_in[7];
    const float* c1_w2   = (const float*)d_in[8];
    const float* c1_b2   = (const float*)d_in[9];
    const float* c2_w1   = (const float*)d_in[10];
    const float* c2_b1   = (const float*)d_in[11];
    const float* c2_gam  = (const float*)d_in[12];
    const float* c2_bet  = (const float*)d_in[13];
    const float* c2_w2   = (const float*)d_in[14];
    const float* c2_b2   = (const float*)d_in[15];
    const float* g_l1_w  = (const float*)d_in[16];
    const float* g_l1_b  = (const float*)d_in[17];
    const float* g_l2_w  = (const float*)d_in[18];
    const float* g_l2_b  = (const float*)d_in[19];
    const float* fc1_w   = (const float*)d_in[20];
    const float* fc1_b   = (const float*)d_in[21];
    const float* fc2_w   = (const float*)d_in[22];
    const float* fc2_b   = (const float*)d_in[23];
    const float* fc3_w   = (const float*)d_in[24];
    const float* fc3_b   = (const float*)d_in[25];
    const float* fin_w   = (const float*)d_in[26];
    const float* fin_b   = (const float*)d_in[27];

    char* ws = (char*)d_ws;
    size_t off = 0;
    auto walloc = [&](size_t bytes) -> void* {
        void* p = ws + off;
        off += (bytes + 255) & ~(size_t)255;
        return p;
    };
    const size_t NB = (size_t)N_NODES * DIMF * sizeof(float);
    float* bufA      = (float*)walloc(NB);
    float* bufB      = (float*)walloc(NB);
    float* csum      = (float*)walloc(2 * DIMF * sizeof(float));   // csum[128] + csq[128]
    float* csq       = csum + DIMF;
    float* pooled    = (float*)walloc((size_t)N_GRAPHS * DIMF * sizeof(float));
    float* tbuf      = (float*)walloc((size_t)N_GRAPHS * DIMF * sizeof(float));
    float* cbuf1     = (float*)walloc((size_t)N_GRAPHS * DIMF * sizeof(float));
    float* cbuf2     = (float*)walloc((size_t)N_GRAPHS * DIMF * sizeof(float));
    float* logits    = (float*)walloc((size_t)N_GRAPHS * OUTD * sizeof(float));
    float* code_emb  = (float*)walloc((size_t)N_GRAPHS * OUTD * sizeof(float));
    float* trans_emb = (float*)walloc((size_t)N_GRAPHS * OUTD * sizeof(float));
    // packed bf16 weights (fragment order)
    unsigned short* pw_c1w1 = (unsigned short*)walloc((size_t)KDIM * DIMF * 2);
    unsigned short* pw_c1w2 = (unsigned short*)walloc((size_t)KDIM * DIMF * 2);
    unsigned short* pw_c2w1 = (unsigned short*)walloc((size_t)KDIM * DIMF * 2);
    unsigned short* pw_c2w2 = (unsigned short*)walloc((size_t)KDIM * DIMF * 2);
    unsigned short* pw_gl1  = (unsigned short*)walloc((size_t)KDIM * DIMF * 2);
    unsigned short* pw_fc1  = (unsigned short*)walloc((size_t)KDIM * DIMF * 2);
    unsigned short* pw_fc2  = (unsigned short*)walloc((size_t)KDIM * DIMF * 2);
    unsigned short* pw_gl2  = (unsigned short*)walloc((size_t)KDIM * OUTD * 2);
    unsigned short* pw_fc3  = (unsigned short*)walloc((size_t)KDIM * OUTD * 2);
    float* out = (float*)d_out;

    auto pack = [&](const float* W, unsigned short* P, int Nc) {
        pack_weights_bf16<<<(KDIM * Nc + 255) / 256, 256, 0, stream>>>(W, P, Nc);
    };
    pack(c1_w1, pw_c1w1, DIMF); pack(c1_w2, pw_c1w2, DIMF);
    pack(c2_w1, pw_c2w1, DIMF); pack(c2_w2, pw_c2w2, DIMF);
    pack(g_l1_w, pw_gl1, DIMF); pack(fc1_w, pw_fc1, DIMF); pack(fc2_w, pw_fc2, DIMF);
    pack(g_l2_w, pw_gl2, OUTD); pack(fc3_w, pw_fc3, OUTD);

    auto launch_gemm = [&](const float* A, const unsigned short* Wp, const float* b,
                           float* C, int M, int Nc, int act) {
        const int strips = (M + 15) / 16;
        const int blocks = (strips + 7) / 8;       // 8 waves (strips) per block
        if (Nc == DIMF) {
            if (act) gemm_wmma_rowstrip<1, 8><<<blocks, 256, 0, stream>>>(A, Wp, b, C, M);
            else     gemm_wmma_rowstrip<0, 8><<<blocks, 256, 0, stream>>>(A, Wp, b, C, M);
        } else {   // Nc == OUTD, no activation needed in this model
            gemm_wmma_rowstrip<0, 4><<<blocks, 256, 0, stream>>>(A, Wp, b, C, M);
        }
    };

    const int edgeBlocks = (int)(((long long)N_EDGES * 32 + 255) / 256);
    const int nodeBlocks = (int)(((long long)N_NODES * 32 + 255) / 256);
    const int ewBlocks   = (int)(((long long)N_NODES * DIMF + 255) / 256);

    // one GIN conv: hout = relu( relu(BN((hin + agg)@w1 + b1)) @ w2 + b2 )
    auto conv = [&](const float* hin, float* zbuf, float* ybuf, float* hout,
                    const unsigned short* w1, const float* b1,
                    const float* gam, const float* bet,
                    const unsigned short* w2, const float* b2) {
        (void)hipMemcpyAsync(zbuf, hin, NB, hipMemcpyDeviceToDevice, stream); // z = h
        edge_scatter_add<<<edgeBlocks, 256, 0, stream>>>(ei, hin, zbuf);      // z += agg
        launch_gemm(zbuf, w1, b1, ybuf, N_NODES, DIMF, 0);                    // y = z@w1+b1
        fill_zero<<<1, 256, 0, stream>>>(csum, 2 * DIMF);
        column_stats<<<512, DIMF, 0, stream>>>(ybuf, csum, csq, N_NODES);
        bn_relu<<<ewBlocks, 256, 0, stream>>>(ybuf, csum, csq, gam, bet, N_NODES);
        launch_gemm(ybuf, w2, b2, hout, N_NODES, DIMF, 1);                    // h = relu(y@w2+b2)
    };

    // GIN branch
    conv(x,    bufA, bufB, bufA, pw_c1w1, c1_b1, c1_gam, c1_bet, pw_c1w2, c1_b2);  // h1 -> bufA
    conv(bufA, bufB, bufA, bufB, pw_c2w1, c2_b1, c2_gam, c2_bet, pw_c2w2, c2_b2);  // h2 -> bufB

    fill_zero<<<(int)(((long long)N_GRAPHS * DIMF + 255) / 256), 256, 0, stream>>>(
        pooled, (long long)N_GRAPHS * DIMF);
    pool_scatter_add<<<nodeBlocks, 256, 0, stream>>>(batch, bufB, pooled);

    launch_gemm(pooled, pw_gl1, g_l1_b, tbuf,      N_GRAPHS, DIMF, 1);
    launch_gemm(tbuf,   pw_gl2, g_l2_b, trans_emb, N_GRAPHS, OUTD, 0);

    // code (MLP) branch
    launch_gemm(code_x, pw_fc1, fc1_b, cbuf1,  N_GRAPHS, DIMF, 1);
    launch_gemm(cbuf1,  pw_fc2, fc2_b, cbuf2,  N_GRAPHS, DIMF, 1);
    launch_gemm(cbuf2,  pw_fc3, fc3_b, logits, N_GRAPHS, OUTD, 0);
    log_softmax_64<<<N_GRAPHS / 8, 256, 0, stream>>>(logits, code_emb);

    // fusion head
    final_head<<<(N_GRAPHS + 255) / 256, 256, 0, stream>>>(code_emb, trans_emb,
                                                           fin_w, fin_b, out);
}
